// RNNLayer_10445360464552
// MI455X (gfx1250) — matrix operand order repaired and checked by reading, not compiled, not run on hardware
//
#include <hip/hip_runtime.h>
#include <hip/hip_bf16.h>
#include <math.h>

// ---------------------------------------------------------------------------
// Quantized LSTM forward for MI455X (gfx1250, wave32, WMMA f32_16x16x32_f16)
// B=64, T=128, I=512, H=1024.
// Workspace: ~20.5 MB base; +128 MB optional Gx = x@Wx precompute (used when
// ws_size permits -> recurrent critical path shrinks from 48 to 32 K-chunks).
// Step kernel: 4 waves/tile split the K dimension, LDS-reduce partials,
// prefetch-1 software pipelining enforced with sched_barrier.
// ---------------------------------------------------------------------------

typedef __attribute__((ext_vector_type(16))) _Float16 v16h;
typedef __attribute__((ext_vector_type(8)))  float    v8f;

#define B_   64
#define T_   128
#define I_   512
#define H_   1024
#define NG_  4096      // 4*H
#define NT_  256       // NG/16 column tiles
#define KCX_ 16        // I/32 k-chunks
#define KCH_ 32        // H/32 k-chunks
#define RT_  4         // B/16 row tiles
#define WV_  4         // waves per step-kernel block (K-split factor)

#if __has_builtin(__builtin_amdgcn_sched_barrier)
#define SCHED_FENCE() __builtin_amdgcn_sched_barrier(0)
#else
#define SCHED_FENCE()
#endif

// ---- monotonic float <-> uint encoding for atomic min/max -----------------
__device__ __forceinline__ unsigned fenc(float f) {
  unsigned b = __float_as_uint(f);
  return (b & 0x80000000u) ? ~b : (b | 0x80000000u);
}
__device__ __forceinline__ float fdec(unsigned u) {
  unsigned b = (u & 0x80000000u) ? (u & 0x7fffffffu) : ~u;
  return __uint_as_float(b);
}
__device__ __forceinline__ unsigned umin2(unsigned a, unsigned b){ return a < b ? a : b; }
__device__ __forceinline__ unsigned umax2(unsigned a, unsigned b){ return a > b ? a : b; }

// fake-quant: clip(round(x/s)+z, qmin, qmax); dequant back
__device__ __forceinline__ float fq(float x, float s, float z, float qmn, float qmx) {
  float q = fminf(fmaxf(rintf(x / s) + z, qmn), qmx);
  return (q - z) * s;
}

#define WMMA_F16(A, B, C) \
  __builtin_amdgcn_wmma_f32_16x16x32_f16(false, (A), false, (B), (short)0, (C), false, false)

// ---------------------------------------------------------------------------
__global__ void k_init_slots(unsigned* slots) {
  int i = threadIdx.x;
  if (i < 4) { slots[2 * i] = 0xFFFFFFFFu; slots[2 * i + 1] = 0u; }
}

__global__ void k_minmax(const float* __restrict__ x, long n,
                         unsigned* slots, int slotBase) {
  __shared__ unsigned smn[256], smx[256];
  unsigned mn = 0xFFFFFFFFu, mx = 0u;
  long i = (long)blockIdx.x * blockDim.x + threadIdx.x;
  long stride = (long)gridDim.x * blockDim.x;
  for (; i < n; i += stride) { unsigned e = fenc(x[i]); mn = umin2(mn, e); mx = umax2(mx, e); }
  smn[threadIdx.x] = mn; smx[threadIdx.x] = mx;
  __syncthreads();
  for (int s = 128; s > 0; s >>= 1) {
    if ((int)threadIdx.x < s) {
      smn[threadIdx.x] = umin2(smn[threadIdx.x], smn[threadIdx.x + s]);
      smx[threadIdx.x] = umax2(smx[threadIdx.x], smx[threadIdx.x + s]);
    }
    __syncthreads();
  }
  if (threadIdx.x == 0) {
    atomicMin(&slots[slotBase], smn[0]);
    atomicMax(&slots[slotBase + 1], smx[0]);
  }
}

// qp layout: [0]in_s [1]in_z [2]h_s [3]h_z [4]c_s [5]c_z [6]o_s [7]o_z [8]qmin [9]qmax
__global__ void k_qparams(const unsigned* __restrict__ slots,
                          const int* __restrict__ qbits,
                          float* __restrict__ qp, int which) {
  if (threadIdx.x != 0) return;
  int bits = qbits[0];
  float qmn = -(float)(1 << (bits - 1));
  float qmx = (float)((1 << (bits - 1)) - 1);
  if (which == 0) {
    qp[8] = qmn; qp[9] = qmx;
    for (int t = 0; t < 3; t++) {
      float mn = fminf(fdec(slots[2 * t]), 0.f);
      float mx = fmaxf(fdec(slots[2 * t + 1]), 0.f);
      float s = fmaxf((mx - mn) / (qmx - qmn), 1e-8f);
      float z = fminf(fmaxf(rintf(qmn - mn / s), qmn), qmx);
      qp[2 * t] = s; qp[2 * t + 1] = z;
    }
  } else {
    float mn = fminf(fdec(slots[6]), 0.f);
    float mx = fmaxf(fdec(slots[7]), 0.f);
    float s = fmaxf((mx - mn) / (qmx - qmn), 1e-8f);
    float z = fminf(fmaxf(rintf(qmn - mn / s), qmn), qmx);
    qp[6] = s; qp[7] = z;
  }
}

// ---- pack inputs: fake-quant -> f16 A-fragment layout [T][RT][KCX][lane32][16]
// A 16x32 f16 layout: lane = m + 16*((k>>3)&1), f16 slot j = (k&7) + 8*(k>>4)
__global__ void k_pack_x(const float* __restrict__ x, _Float16* __restrict__ xq,
                         const float* __restrict__ qp) {
  float s = qp[0], z = qp[1], qmn = qp[8], qmx = qp[9];
  long n = (long)B_ * T_ * I_;
  for (long idx = (long)blockIdx.x * blockDim.x + threadIdx.x; idx < n;
       idx += (long)gridDim.x * blockDim.x) {
    long b = idx / ((long)T_ * I_);
    long r = idx % ((long)T_ * I_);
    long t = r / I_;
    long i = r % I_;
    float v = fq(x[idx], s, z, qmn, qmx);
    int rt = (int)(b >> 4), m = (int)(b & 15);
    int kc = (int)(i >> 5), kk = (int)(i & 31);
    int lane = m + 16 * ((kk >> 3) & 1);
    int j = (kk & 7) + 8 * (kk >> 4);
    long dst = ((((long)t * RT_ + rt) * KCX_ + kc) * 32 + lane) * 16 + j;
    xq[dst] = (_Float16)v;
  }
}

// ---- pack weights: f32 [K, 4H] row-major -> f16 B-fragment layout [K/32][NT][lane32][16]
// B 32x16 f16 layout: lane = n + 16*((k>>3)&1), slot j = (k&7) + 8*(k>>4)
__global__ void k_pack_w(const float* __restrict__ w, _Float16* __restrict__ wp, int K) {
  long n = (long)K * NG_;
  for (long idx = (long)blockIdx.x * blockDim.x + threadIdx.x; idx < n;
       idx += (long)gridDim.x * blockDim.x) {
    long k = idx / NG_;
    long c = idx % NG_;
    int kc = (int)(k >> 5), kk = (int)(k & 31);
    int nt = (int)(c >> 4), nn = (int)(c & 15);
    int lane = nn + 16 * ((kk >> 3) & 1);
    int j = (kk & 7) + 8 * (kk >> 4);
    long dst = (((long)kc * NT_ + nt) * 32 + lane) * 16 + j;
    wp[dst] = (_Float16)w[idx];
  }
}

// ---- pack initial h (fake-quant) into A-fragment layout [RT][KCH][lane32][16]
__global__ void k_pack_h0(const float* __restrict__ h, _Float16* __restrict__ hq,
                          const float* __restrict__ qp) {
  float s = qp[2], z = qp[3], qmn = qp[8], qmx = qp[9];
  long n = (long)B_ * H_;
  for (long idx = (long)blockIdx.x * blockDim.x + threadIdx.x; idx < n;
       idx += (long)gridDim.x * blockDim.x) {
    int b = (int)(idx / H_), hh = (int)(idx % H_);
    int rt = b >> 4, m = b & 15;
    int kc = hh >> 5, kk = hh & 31;
    int lane = m + 16 * ((kk >> 3) & 1);
    int j = (kk & 7) + 8 * (kk >> 4);
    long dst = (((long)rt * KCH_ + kc) * 32 + lane) * 16 + j;
    hq[dst] = (_Float16)fq(h[idx], s, z, qmn, qmx);
  }
}

// ---- c kept unquantized f32, transposed [H][B] so epilogue reads are contiguous
__global__ void k_pack_c(const float* __restrict__ c, float* __restrict__ ct) {
  long n = (long)B_ * H_;
  for (long idx = (long)blockIdx.x * blockDim.x + threadIdx.x; idx < n;
       idx += (long)gridDim.x * blockDim.x) {
    int b = (int)(idx / H_), hh = (int)(idx % H_);
    ct[(long)hh * B_ + b] = c[idx];
  }
}

// ---------------------------------------------------------------------------
// Pipelined 4-gate K-segment: accumulate chunks [kc0, kc1) of aBase x wBase
// into a0..a3.  Prefetch distance 1, enforced with a scheduling barrier so
// the compiler cannot sink the next-chunk loads down to their WMMA uses
// (observed: without the fence the B loads degrade to s_wait_loadcnt 0x0).
// ---------------------------------------------------------------------------
__device__ __forceinline__ void gemm_seg(const v16h* __restrict__ aBase,   // + kc*32 per chunk (lane folded in)
                                         const v16h* __restrict__ wBase,   // weight fragments
                                         int nt, int lane, int kc0, int kc1,
                                         v8f& a0, v8f& a1, v8f& a2, v8f& a3) {
  v16h an  = aBase[(long)kc0 * 32];
  v16h b0n = wBase[((long)kc0 * NT_ + 0 * 64 + nt) * 32 + lane];
  v16h b1n = wBase[((long)kc0 * NT_ + 1 * 64 + nt) * 32 + lane];
  v16h b2n = wBase[((long)kc0 * NT_ + 2 * 64 + nt) * 32 + lane];
  v16h b3n = wBase[((long)kc0 * NT_ + 3 * 64 + nt) * 32 + lane];
  for (int kc = kc0; kc < kc1; kc++) {
    v16h a = an, b0 = b0n, b1 = b1n, b2 = b2n, b3 = b3n;
    int kn = kc + 1;
    if (kn < kc1) {
      an  = aBase[(long)kn * 32];
      b0n = wBase[((long)kn * NT_ + 0 * 64 + nt) * 32 + lane];
      b1n = wBase[((long)kn * NT_ + 1 * 64 + nt) * 32 + lane];
      b2n = wBase[((long)kn * NT_ + 2 * 64 + nt) * 32 + lane];
      b3n = wBase[((long)kn * NT_ + 3 * 64 + nt) * 32 + lane];
    }
    SCHED_FENCE();                 // keep next-chunk loads above these WMMAs
    a0 = WMMA_F16(a, b0, a0);
    a1 = WMMA_F16(a, b1, a1);
    a2 = WMMA_F16(a, b2, a2);
    a3 = WMMA_F16(a, b3, a3);
    SCHED_FENCE();
  }
}

// ---------------------------------------------------------------------------
// Hoisted time-parallel GEMM: Gx[t] = x_t @ Wx for all t at once.
// grid = (H/16, B/16, T), one wave per 16x16x4-gate tile set; 32K independent
// tile-waves saturate the WMMA units.  Output stored as C-layout fragments
// [t][rt][gtile(256)][lane32][8 f32].
// ---------------------------------------------------------------------------
__global__ void __launch_bounds__(32)
k_gx(const _Float16* __restrict__ xq, const _Float16* __restrict__ wxp,
     float* __restrict__ gx) {
  const int lane = threadIdx.x;
  const int nt = blockIdx.x;   // 0..63
  const int rt = blockIdx.y;   // 0..3
  const int t  = blockIdx.z;   // 0..127

  v8f a0 = {}, a1 = {}, a2 = {}, a3 = {};
  const v16h* xt = (const v16h*)xq + (((long)t * RT_ + rt) * KCX_) * 32 + lane;
  gemm_seg(xt, (const v16h*)wxp, nt, lane, 0, KCX_, a0, a1, a2, a3);

  long base = (((long)t * RT_ + rt) * NT_);
  *(v8f*)(gx + ((base + 0 * 64 + nt) * 32 + lane) * 8) = a0;
  *(v8f*)(gx + ((base + 1 * 64 + nt) * 32 + lane) * 8) = a1;
  *(v8f*)(gx + ((base + 2 * 64 + nt) * 32 + lane) * 8) = a2;
  *(v8f*)(gx + ((base + 3 * 64 + nt) * 32 + lane) * 8) = a3;
}

// ---------------------------------------------------------------------------
// One LSTM step.  grid = (H/16, B/16), block = 128 (4 waves per 16x16 tile).
// The K dimension is split across the 4 waves (8 recurrent chunks each, plus
// 4 input chunks each on the fallback path), partials LDS-reduced with one
// barrier, epilogue on wave 0.  launch_bounds(128,1): only ~1 block/WGP runs,
// so spend VGPRs on the pipelined fragment set rather than on occupancy.
// ---------------------------------------------------------------------------
__global__ void __launch_bounds__(128, 1)
k_step(const _Float16* __restrict__ xq, const _Float16* __restrict__ wxp,
       const _Float16* __restrict__ whp, const float* __restrict__ bias,
       const _Float16* __restrict__ hq_in, _Float16* __restrict__ hq_out,
       float* __restrict__ ct, float* __restrict__ out,
       float* __restrict__ hlast, float* __restrict__ clast,
       const float* __restrict__ qp, const float* __restrict__ gx,
       int t, int isLast, int useGx) {
  const int lane = threadIdx.x & 31;
  const int w    = threadIdx.x >> 5;   // 0..3 : K-split wave id
  const int nt = blockIdx.x;           // 0..63  column tile over H
  const int rt = blockIdx.y;           // 0..3   row tile over B

  __shared__ __align__(32) float red[WV_][32][32];   // 16 KB partial accs

  v8f a0 = {}, a1 = {}, a2 = {}, a3 = {};
  if (useGx) {
    if (w == 0) {
      long base = (((long)t * RT_ + rt) * NT_);
      a0 = *(const v8f*)(gx + ((base + 0 * 64 + nt) * 32 + lane) * 8);
      a1 = *(const v8f*)(gx + ((base + 1 * 64 + nt) * 32 + lane) * 8);
      a2 = *(const v8f*)(gx + ((base + 2 * 64 + nt) * 32 + lane) * 8);
      a3 = *(const v8f*)(gx + ((base + 3 * 64 + nt) * 32 + lane) * 8);
    }
  } else {
    const v16h* xt = (const v16h*)xq + (((long)t * RT_ + rt) * KCX_) * 32 + lane;
    gemm_seg(xt, (const v16h*)wxp, nt, lane,
             w * (KCX_ / WV_), (w + 1) * (KCX_ / WV_), a0, a1, a2, a3);
  }

  const v16h* ht = (const v16h*)hq_in + ((long)rt * KCH_) * 32 + lane;
  gemm_seg(ht, (const v16h*)whp, nt, lane,
           w * (KCH_ / WV_), (w + 1) * (KCH_ / WV_), a0, a1, a2, a3);

  // ---- cross-wave K reduction through LDS ----
  if (w != 0) {
    *(v8f*)&red[w][lane][0]  = a0;
    *(v8f*)&red[w][lane][8]  = a1;
    *(v8f*)&red[w][lane][16] = a2;
    *(v8f*)&red[w][lane][24] = a3;
  }
  __syncthreads();
  if (w != 0) return;
#pragma unroll
  for (int ww = 1; ww < WV_; ww++) {
    a0 = a0 + *(const v8f*)&red[ww][lane][0];
    a1 = a1 + *(const v8f*)&red[ww][lane][8];
    a2 = a2 + *(const v8f*)&red[ww][lane][16];
    a3 = a3 + *(const v8f*)&red[ww][lane][24];
  }

  // ---- fused LSTM cell epilogue (C layout: lane->N, VGPR v->M) ----
  const int nn = lane & 15;
  const int hcol = nt * 16 + nn;          // 0..1023
  const int mbase = 8 * (lane >> 4);      // 0 or 8
  const float bi = bias[0 * H_ + hcol];
  const float bf = bias[1 * H_ + hcol];
  const float bg = bias[2 * H_ + hcol];
  const float bo = bias[3 * H_ + hcol];
  const float hs = qp[2], hz = qp[3], cs = qp[4], cz = qp[5];
  const float qmn = qp[8], qmx = qp[9];
  float* cp = ct + (long)hcol * B_ + rt * 16 + mbase;   // 8 contiguous f32

  const int kc = hcol >> 5, kk = hcol & 31;
  const int laneA = 16 * ((kk >> 3) & 1);               // +m added per element
  const int jA = (kk & 7) + 8 * (kk >> 4);

#pragma unroll
  for (int v = 0; v < 8; v++) {
    float ig = 1.f / (1.f + expf(-(a0[v] + bi)));
    float fg = 1.f / (1.f + expf(-(a1[v] + bf)));
    float gg = tanhf(a2[v] + bg);
    float og = 1.f / (1.f + expf(-(a3[v] + bo)));
    float cq = fq(cp[v], cs, cz, qmn, qmx);             // per-step SimQuant of c
    float cn = fg * cq + ig * gg;
    float hn = og * tanhf(cn);
    cp[v] = cn;                                         // carry c unquantized
    int m = mbase + v;
    int brow = rt * 16 + m;
    out[((long)brow * T_ + t) * H_ + hcol] = hn;        // outputs unquantized here
    // pre-quantize h for next step's A fragment
    float hqv = fq(hn, hs, hz, qmn, qmx);
    hq_out[(((long)rt * KCH_ + kc) * 32 + (m + laneA)) * 16 + jA] = (_Float16)hqv;
    if (isLast) {
      hlast[(long)brow * H_ + hcol] = hn;
      clast[(long)brow * H_ + hcol] = cn;
    }
  }
}

// ---- final SimQuant of outputs in place -----------------------------------
__global__ void k_fq_out(float* __restrict__ out, long n, const float* __restrict__ qp) {
  float s = qp[6], z = qp[7], qmn = qp[8], qmx = qp[9];
  for (long i = (long)blockIdx.x * blockDim.x + threadIdx.x; i < n;
       i += (long)gridDim.x * blockDim.x)
    out[i] = fq(out[i], s, z, qmn, qmx);
}

// ---------------------------------------------------------------------------
extern "C" void kernel_launch(void* const* d_in, const int* in_sizes, int n_in,
                              void* d_out, int out_size, void* d_ws, size_t ws_size,
                              hipStream_t stream) {
  const float* inputs = (const float*)d_in[0];
  const float* h_prev = (const float*)d_in[1];
  const float* c_prev = (const float*)d_in[2];
  const float* Wx     = (const float*)d_in[3];
  const float* Wh     = (const float*)d_in[4];
  const float* bias   = (const float*)d_in[5];
  const int*   qbits  = (const int*)d_in[6];

  float* out   = (float*)d_out;                       // [B,T,H] fake-quanted
  float* hlast = out + (long)B_ * T_ * H_;            // [B,H] unquantized
  float* clast = hlast + (long)B_ * H_;               // [B,H] unquantized

  char* ws = (char*)d_ws;
  unsigned* slots = (unsigned*)ws;                                   // 32 B
  float*    qp    = (float*)(ws + 64);                               // 40 B
  size_t off = 256;
  _Float16* xq  = (_Float16*)(ws + off); off += (size_t)T_ * B_ * I_ * 2;   // 8 MB
  _Float16* wxp = (_Float16*)(ws + off); off += (size_t)I_ * NG_ * 2;       // 4 MB
  _Float16* whp = (_Float16*)(ws + off); off += (size_t)H_ * NG_ * 2;       // 8 MB
  _Float16* hq0 = (_Float16*)(ws + off); off += (size_t)B_ * H_ * 2;        // 128 KB
  _Float16* hq1 = (_Float16*)(ws + off); off += (size_t)B_ * H_ * 2;        // 128 KB
  float*    ct  = (float*)(ws + off);    off += (size_t)B_ * H_ * 4;        // 256 KB
  float*    gx  = (float*)(ws + off);
  size_t gxBytes = (size_t)T_ * B_ * NG_ * 4;                               // 128 MB
  int useGx = (ws_size >= off + gxBytes) ? 1 : 0;

  // 1) qparams for inputs / h_prev / c_prev
  k_init_slots<<<1, 32, 0, stream>>>(slots);
  k_minmax<<<512, 256, 0, stream>>>(inputs, (long)B_ * T_ * I_, slots, 0);
  k_minmax<<<64, 256, 0, stream>>>(h_prev, (long)B_ * H_, slots, 2);
  k_minmax<<<64, 256, 0, stream>>>(c_prev, (long)B_ * H_, slots, 4);
  k_qparams<<<1, 1, 0, stream>>>(slots, qbits, qp, 0);

  // 2) one-time packing into WMMA fragment layouts
  k_pack_x<<<2048, 256, 0, stream>>>(inputs, xq, qp);
  k_pack_w<<<1024, 256, 0, stream>>>(Wx, wxp, I_);
  k_pack_w<<<2048, 256, 0, stream>>>(Wh, whp, H_);
  k_pack_h0<<<64, 256, 0, stream>>>(h_prev, hq0, qp);
  k_pack_c<<<64, 256, 0, stream>>>(c_prev, ct);

  // 2b) hoist the time-parallel x@Wx GEMM off the serial critical path
  if (useGx)
    k_gx<<<dim3(H_ / 16, B_ / 16, T_), 32, 0, stream>>>(xq, wxp, gx);

  // 3) sequential scan, ping-pong the quantized-h fragment buffers
  for (int t = 0; t < T_; t++) {
    const _Float16* hin = (t & 1) ? hq1 : hq0;
    _Float16*       hout = (t & 1) ? hq0 : hq1;
    k_step<<<dim3(H_ / 16, B_ / 16), WV_ * 32, 0, stream>>>(
        xq, wxp, whp, bias, hin, hout, ct, out, hlast, clast, qp, gx, t,
        (t == T_ - 1) ? 1 : 0, useGx);
  }

  // 4) output qparams + in-place fake-quant of outputs
  k_minmax<<<512, 256, 0, stream>>>(out, (long)B_ * T_ * H_, slots, 6);
  k_qparams<<<1, 1, 0, stream>>>(slots, qbits, qp, 1);
  k_fq_out<<<1024, 256, 0, stream>>>(out, (long)B_ * T_ * H_, qp);
}
